// DeepEPMoE_6640019439878
// MI455X (gfx1250) — matrix operand order repaired and checked by reading, not compiled, not run on hardware
//
#include <hip/hip_runtime.h>
#include <hip/hip_bf16.h>
#include <math.h>

// ---------------------------------------------------------------------------
// MoE (top-2) fused MLP for MI455X (gfx1250), wave32 + WMMA bf16.
//
// T=4096 tokens, D=1024, H=4096, E=8, top-2 => 137 GFLOP (compute-bound).
// v_wmma_f32_16x16x32_bf16 for both GEMMs; weights pre-converted to bf16
// (128MB, fits 192MB L2) and pre-transposed so B-fragments are contiguous
// along K. W-chunk staging via global_load_async_to_lds_b128 (ASYNCcnt);
// next-chunk W1 staging overlaps fc2 compute. Fragment loads are 2-deep
// software-pipelined around the WMMAs.
// ---------------------------------------------------------------------------

typedef __attribute__((ext_vector_type(16))) __bf16 v16bf;
typedef __attribute__((ext_vector_type(8)))  __bf16 v8bf;
typedef __attribute__((ext_vector_type(8)))  float  v8f;

#define T_TOK 4096
#define DDIM  1024
#define HDIM  4096
#define NEXP  8
#define TILE_M 64
#define NWAVES 16
#define BLK   (NWAVES * 32)

// LDS pitches (elements), padded to avoid 64-bank stride conflicts
#define XP   1032   // X tile rows (bf16)
#define W1P  1032   // W1 chunk rows: [32 h'][1024 d]
#define W2P  40     // W2 chunk rows: [1024 d][32 h']
#define HP   40     // Htile rows:    [64 m][32 h']

// LDS byte offsets (all 16B aligned)
#define OFF_X    0
#define OFF_W1   132096                    // 64*1032*2
#define OFF_W2   (OFF_W1 + 32*W1P*2)       // +66048  = 198144
#define OFF_H    (OFF_W2 + 1024*W2P*2)     // +81920  = 280064
#define OFF_RED  (OFF_H  + 64*HP*2)        // +5120   = 285184
#define OFF_TOK  (OFF_RED + 8*288*4)       // +9216   = 294400
#define OFF_CW   (OFF_TOK + 64*4)          // +256    = 294656
#define LDS_BYTES (OFF_CW + 64*4)          //          = 294912

__device__ __forceinline__ v16bf make_ab(v8bf lo, v8bf hi) {
  union { v16bf v; v8bf h[2]; } u;
  u.h[0] = lo; u.h[1] = hi;
  return u.v;
}

// async 16B copy: global -> LDS via the gfx1250 async data path (ASYNCcnt).
// GV mode: per-lane 64-bit global address, per-lane 32-bit LDS offset.
__device__ __forceinline__ void async_copy16(unsigned ldsoff,
                                             unsigned long long gaddr) {
  asm volatile("global_load_async_to_lds_b128 %0, %1, off"
               :: "v"(ldsoff), "v"(gaddr) : "memory");
}

__device__ __forceinline__ void wait_async0() {
#if defined(__has_builtin)
#if __has_builtin(__builtin_amdgcn_s_wait_asynccnt)
  __builtin_amdgcn_s_wait_asynccnt(0);
#else
  asm volatile("s_wait_asynccnt 0x0" ::: "memory");
#endif
#else
  asm volatile("s_wait_asynccnt 0x0" ::: "memory");
#endif
}

// ---------------------------------------------------------------------------
// Kernel 1: tiled transpose + fp32 -> bf16 convert.
// src: [E][R][C] fp32 ; dst: [E][C][R] bf16
// ---------------------------------------------------------------------------
__global__ __launch_bounds__(256) void transpose_cvt(
    __bf16* __restrict__ dst, const float* __restrict__ src, int R, int C) {
  __shared__ float tile[32][33];
  const size_t batch = (size_t)blockIdx.z * (size_t)R * (size_t)C;
  const int c0 = blockIdx.x * 32, r0 = blockIdx.y * 32;
  const int tx = threadIdx.x & 31, ty = threadIdx.x >> 5;   // 32x8
#pragma unroll
  for (int i = 0; i < 32; i += 8)
    tile[ty + i][tx] = src[batch + (size_t)(r0 + ty + i) * C + c0 + tx];
  __syncthreads();
#pragma unroll
  for (int i = 0; i < 32; i += 8)
    dst[batch + (size_t)(c0 + ty + i) * R + r0 + tx] = (__bf16)tile[tx][ty + i];
}

// ---------------------------------------------------------------------------
// Kernel 2: router. One wave per token. softmax -> top2 -> renormalize.
// ---------------------------------------------------------------------------
__global__ __launch_bounds__(256) void moe_router(
    const float* __restrict__ x, const float* __restrict__ rw,
    float* __restrict__ cw, int* __restrict__ counts, int* __restrict__ lists) {
  __shared__ float rws[NEXP][DDIM];
  for (int i = threadIdx.x; i < NEXP * DDIM; i += 256)
    ((float*)rws)[i] = rw[i];
  __syncthreads();

  const int lane = threadIdx.x & 31;
  const int wv   = threadIdx.x >> 5;
  const int tok  = blockIdx.x * 8 + wv;

  float a[NEXP];
#pragma unroll
  for (int e = 0; e < NEXP; ++e) a[e] = 0.0f;

  const float* xr = x + (size_t)tok * DDIM;
  for (int k = lane; k < DDIM; k += 32) {
    const float xv = xr[k];
#pragma unroll
    for (int e = 0; e < NEXP; ++e) a[e] += xv * rws[e][k];
  }
#pragma unroll
  for (int e = 0; e < NEXP; ++e)
    for (int off = 16; off; off >>= 1) a[e] += __shfl_xor(a[e], off, 32);

  if (lane == 0) {
    float mx = a[0];
#pragma unroll
    for (int e = 1; e < NEXP; ++e) mx = fmaxf(mx, a[e]);
    float p[NEXP], den = 0.0f;
#pragma unroll
    for (int e = 0; e < NEXP; ++e) { p[e] = __expf(a[e] - mx); den += p[e]; }
    const float inv = 1.0f / den;
#pragma unroll
    for (int e = 0; e < NEXP; ++e) p[e] *= inv;
    int e1 = 0;
#pragma unroll
    for (int e = 1; e < NEXP; ++e) if (p[e] > p[e1]) e1 = e;
    int e2 = (e1 == 0) ? 1 : 0;
#pragma unroll
    for (int e = 0; e < NEXP; ++e) if (e != e1 && p[e] > p[e2]) e2 = e;
    const float s  = p[e1] + p[e2] + 1e-9f;
    const float g1 = p[e1] / s, g2 = p[e2] / s;
    float row[NEXP];
#pragma unroll
    for (int e = 0; e < NEXP; ++e) row[e] = 0.0f;
    row[e1] = g1; row[e2] = g2;
#pragma unroll
    for (int e = 0; e < NEXP; ++e) cw[(size_t)tok * NEXP + e] = row[e];
    int s1 = atomicAdd(&counts[e1], 1); lists[e1 * T_TOK + s1] = tok;
    int s2 = atomicAdd(&counts[e2], 1); lists[e2 * T_TOK + s2] = tok;
  }
}

// ---------------------------------------------------------------------------
// Kernel 3: fused expert MLP.  block = (expert e, 64-token tile).
// ---------------------------------------------------------------------------
__global__ __launch_bounds__(BLK) void moe_mlp(
    const float* __restrict__ x, const __bf16* __restrict__ w1t,
    const __bf16* __restrict__ w2t, const float* __restrict__ cw,
    const int* __restrict__ counts, const int* __restrict__ lists,
    float* __restrict__ out) {
  extern __shared__ char smem[];
  __bf16* Xs   = (__bf16*)(smem + OFF_X);
  __bf16* W1s  = (__bf16*)(smem + OFF_W1);
  __bf16* W2s  = (__bf16*)(smem + OFF_W2);
  __bf16* Hs   = (__bf16*)(smem + OFF_H);
  float*  red  = (float*)(smem + OFF_RED);
  int*    tokid= (int*)(smem + OFF_TOK);
  float*  cwv  = (float*)(smem + OFF_CW);

  const int e  = blockIdx.y;
  const int n  = counts[e];
  const int m0 = blockIdx.x * TILE_M;
  if (m0 >= n) return;

  const int tid  = threadIdx.x;
  const int lane = tid & 31;
  const int wid  = tid >> 5;
  const int nl   = lane & 15;
  const int hi   = lane >> 4;        // lane half
  const int koffA = hi * 16;         // A-fragment K offset pattern
  const int koffB = hi * 16;         // B-fragment K offset pattern

  const __bf16* w1base = w1t + (size_t)e * HDIM * DDIM;
  const __bf16* w2base = w2t + (size_t)e * DDIM * HDIM;

  // async staging helpers (16B per lane per issue; 8 issues = 64KB chunk)
  auto stage_w1 = [&](int hcn) {
    const unsigned long long g1 =
        (unsigned long long)(w1base + (size_t)hcn * DDIM);
#pragma unroll
    for (int i = 0; i < 8; ++i) {
      const int idx = tid + i * BLK;
      const int row = idx >> 7, c8 = idx & 127;        // 128 x 16B per row
      async_copy16((unsigned)(uintptr_t)(W1s + row * W1P + c8 * 8),
                   g1 + ((unsigned long long)row * DDIM + c8 * 8) * 2ull);
    }
  };
  auto stage_w2 = [&](int hcn) {
    const unsigned long long g2 = (unsigned long long)(w2base + hcn);
#pragma unroll
    for (int i = 0; i < 8; ++i) {
      const int idx = tid + i * BLK;
      const int d = idx >> 2, c2 = idx & 3;            // 4 x 16B per row
      async_copy16((unsigned)(uintptr_t)(W2s + d * W2P + c2 * 8),
                   g2 + ((unsigned long long)d * HDIM + c2 * 8) * 2ull);
    }
  };

  // token ids + combine weights for this tile
  if (tid < TILE_M) {
    const int gr = m0 + tid;
    const bool valid = gr < n;
    const int t = lists[e * T_TOK + (valid ? gr : 0)];
    tokid[tid] = t;
    cwv[tid] = valid ? cw[(size_t)t * NEXP + e] : 0.0f;
  }
  __syncthreads();

  // stage X tile: fp32 global -> bf16 LDS (row pitch XP)
  for (int idx = tid; idx < TILE_M * DDIM / 4; idx += BLK) {
    const int m  = idx >> 8;          // 256 float4 per row
    const int c4 = idx & 255;
    const float4 xv = ((const float4*)(x + (size_t)tokid[m] * DDIM))[c4];
    union { __bf16 b[4]; uint2 u; } p;
    p.b[0] = (__bf16)xv.x; p.b[1] = (__bf16)xv.y;
    p.b[2] = (__bf16)xv.z; p.b[3] = (__bf16)xv.w;
    *(uint2*)(Xs + m * XP + c4 * 4) = p.u;
  }
  __syncthreads();

  // prologue: first W1 chunk in flight
  stage_w1(0);

  // persistent fc2 accumulators: wave (mt2, nc) owns rows mt2*16.., cols nc*256..
  const int mt2 = wid & 3, nc = wid >> 2;
  const v8f vzero = {0.f, 0.f, 0.f, 0.f, 0.f, 0.f, 0.f, 0.f};
  v8f acc[16];
#pragma unroll
  for (int j = 0; j < 16; ++j) acc[j] = vzero;

  // fc1 wave roles: tile (mt,nt) of Htile, kh = K half
  const int tid8 = wid & 7, kh = wid >> 3;
  const int mt = tid8 & 3, nt = tid8 >> 2;
  const __bf16* arow1 = Xs  + (mt * 16 + nl) * XP;
  const __bf16* brow1 = W1s + (nt * 16 + nl) * W1P;

#pragma unroll 1
  for (int hc = 0; hc < HDIM; hc += 32) {
    // ---- stage W2 chunk (W1 chunk already in flight from prev iter) ----
    stage_w2(hc);
    if (hc + 32 < HDIM) {   // warm L2/WGP$ for next chunk
      const char* p1 = (const char*)(w1base + (size_t)(hc + 32) * DDIM);
      __builtin_prefetch(p1 + (size_t)tid * 128, 0, 3);        // 512 x 128B
      const char* p2 = (const char*)(w2base + (hc + 32));
      __builtin_prefetch(p2 + (size_t)(tid * 2) * HDIM * 2, 0, 3);
      __builtin_prefetch(p2 + (size_t)(tid * 2 + 1) * HDIM * 2, 0, 3);
    }
    wait_async0();          // covers this wave's W1(hc) and W2(hc) copies
    __syncthreads();

    // ---- fc1: c1[16x16] = X(mt) @ W1(nt); 2-deep pipelined fragments ----
    v8f c1 = vzero;
    const int kbase = kh * 512;
    v8bf a0lo = *(const v8bf*)(arow1 + kbase + koffA);
    v8bf a0hi = *(const v8bf*)(arow1 + kbase + 8 + koffA);
    v8bf b0lo = *(const v8bf*)(brow1 + kbase + koffB);
    v8bf b0hi = *(const v8bf*)(brow1 + kbase + koffB + 8);
    v8bf a1lo = *(const v8bf*)(arow1 + kbase + 32 + koffA);
    v8bf a1hi = *(const v8bf*)(arow1 + kbase + 40 + koffA);
    v8bf b1lo = *(const v8bf*)(brow1 + kbase + 32 + koffB);
    v8bf b1hi = *(const v8bf*)(brow1 + kbase + 40 + koffB);
#pragma unroll
    for (int kk = 0; kk < 512; kk += 64) {
      c1 = __builtin_amdgcn_wmma_f32_16x16x32_bf16(
          false, make_ab(a0lo, a0hi), false, make_ab(b0lo, b0hi),
          (short)0, c1, false, false);
      const int kn0 = kbase + ((kk + 64) & 511);   // tail wraps (unused loads)
      a0lo = *(const v8bf*)(arow1 + kn0 + koffA);
      a0hi = *(const v8bf*)(arow1 + kn0 + 8 + koffA);
      b0lo = *(const v8bf*)(brow1 + kn0 + koffB);
      b0hi = *(const v8bf*)(brow1 + kn0 + koffB + 8);
      c1 = __builtin_amdgcn_wmma_f32_16x16x32_bf16(
          false, make_ab(a1lo, a1hi), false, make_ab(b1lo, b1hi),
          (short)0, c1, false, false);
      const int kn1 = kbase + ((kk + 96) & 511);
      a1lo = *(const v8bf*)(arow1 + kn1 + koffA);
      a1hi = *(const v8bf*)(arow1 + kn1 + 8 + koffA);
      b1lo = *(const v8bf*)(brow1 + kn1 + koffB);
      b1hi = *(const v8bf*)(brow1 + kn1 + koffB + 8);
    }
    if (kh) {                 // store partial for cross-wave reduce
      float* rp = red + tid8 * 288 + lane * 9;
#pragma unroll
      for (int r = 0; r < 8; ++r) rp[r] = c1[r];
    }
    __syncthreads();
    if (!kh) {                // reduce + exact gelu + bf16 -> Htile
      const float* rp = red + tid8 * 288 + lane * 9;
#pragma unroll
      for (int r = 0; r < 8; ++r) {
        const float s = c1[r] + rp[r];
        const float g = 0.5f * s * (1.0f + erff(s * 0.70710678118654752f));
        Hs[(mt * 16 + r + hi * 8) * HP + nt * 16 + nl] = (__bf16)g;
      }
    }
    __syncthreads();          // Hs published; W1s now dead

    // overlap: async-stage next chunk's W1 while fc2 computes
    if (hc + 32 < HDIM) stage_w1(hc + 32);

    // ---- fc2: acc[j] += Htile(mt2) @ W2(nc*256 + j*16); 2-deep pipe ----
    const __bf16* ha = Hs + (mt2 * 16 + nl) * HP;
    const v8bf halo = *(const v8bf*)(ha + koffA);
    const v8bf hahi = *(const v8bf*)(ha + 8 + koffA);
    const v16bf afrag = make_ab(halo, hahi);
    const __bf16* brow2 = W2s + (nc * 256 + nl) * W2P;
    v8bf c0lo = *(const v8bf*)(brow2 + koffB);
    v8bf c0hi = *(const v8bf*)(brow2 + koffB + 8);
    v8bf c1lo = *(const v8bf*)(brow2 + 16 * W2P + koffB);
    v8bf c1hi = *(const v8bf*)(brow2 + 16 * W2P + koffB + 8);
#pragma unroll
    for (int j = 0; j < 16; j += 2) {
      acc[j] = __builtin_amdgcn_wmma_f32_16x16x32_bf16(
          false, afrag, false, make_ab(c0lo, c0hi), (short)0, acc[j],
          false, false);
      const int j2 = (((j + 2) & 15) * 16) * W2P;
      c0lo = *(const v8bf*)(brow2 + j2 + koffB);
      c0hi = *(const v8bf*)(brow2 + j2 + koffB + 8);
      acc[j + 1] = __builtin_amdgcn_wmma_f32_16x16x32_bf16(
          false, afrag, false, make_ab(c1lo, c1hi), (short)0, acc[j + 1],
          false, false);
      const int j3 = (((j + 3) & 15) * 16) * W2P;
      c1lo = *(const v8bf*)(brow2 + j3 + koffB);
      c1hi = *(const v8bf*)(brow2 + j3 + koffB + 8);
    }
    __syncthreads();          // protect W2s/Hs before next chunk staging
  }

  // ---- scatter: out[token, d] += cw * acc   (atomic f32) ----
  const int dbase = nc * 256 + nl;
#pragma unroll
  for (int j = 0; j < 16; ++j) {
#pragma unroll
    for (int r = 0; r < 8; ++r) {
      const int m = mt2 * 16 + r + hi * 8;
      const float wv = cwv[m];
      if (wv != 0.0f)
        atomicAdd(&out[(size_t)tokid[m] * DDIM + dbase + j * 16],
                  wv * acc[j][r]);
    }
  }
}

// ---------------------------------------------------------------------------
extern "C" void kernel_launch(void* const* d_in, const int* in_sizes, int n_in,
                              void* d_out, int out_size, void* d_ws, size_t ws_size,
                              hipStream_t stream) {
  const float* x        = (const float*)d_in[0];
  const float* router_w = (const float*)d_in[1];
  const float* w1       = (const float*)d_in[2];
  const float* w2       = (const float*)d_in[3];
  float* out = (float*)d_out;

  // workspace layout (~128.3 MB): cw | counts | lists | w1t(bf16) | w2t(bf16)
  char* ws = (char*)d_ws;
  float*  cw     = (float*)ws;                               // 4096*8*4 = 131072
  int*    counts = (int*)(ws + 131072);                      // 8 ints (pad 256)
  int*    lists  = (int*)(ws + 131072 + 256);                // 8*4096*4 = 131072
  __bf16* w1t    = (__bf16*)(ws + 262400);                   // 64 MB
  __bf16* w2t    = (__bf16*)(ws + 262400 + (size_t)NEXP * DDIM * HDIM * 2);

  hipMemsetAsync(counts, 0, 8 * sizeof(int), stream);
  hipMemsetAsync(out, 0, (size_t)T_TOK * DDIM * sizeof(float), stream);

  // weight convert+transpose: w1[E][D][H]->w1t[E][H][D], w2[E][H][D]->w2t[E][D][H]
  transpose_cvt<<<dim3(HDIM / 32, DDIM / 32, NEXP), 256, 0, stream>>>(w1t, w1, DDIM, HDIM);
  transpose_cvt<<<dim3(DDIM / 32, HDIM / 32, NEXP), 256, 0, stream>>>(w2t, w2, HDIM, DDIM);

  moe_router<<<T_TOK / 8, 256, 0, stream>>>(x, router_w, cw, counts, lists);

  hipFuncSetAttribute(reinterpret_cast<const void*>(moe_mlp),
                      hipFuncAttributeMaxDynamicSharedMemorySize, LDS_BYTES);
  moe_mlp<<<dim3(T_TOK / TILE_M, NEXP), BLK, LDS_BYTES, stream>>>(
      x, w1t, w2t, cw, counts, lists, out);
}
